// Attention_89197880803737
// MI455X (gfx1250) — matrix operand order repaired and checked by reading, not compiled, not run on hardware
//
#include <hip/hip_runtime.h>

typedef long long i64;
typedef __attribute__((ext_vector_type(16))) __bf16 v16bf;
typedef __attribute__((ext_vector_type(8)))  __bf16 v8bf;
typedef __attribute__((ext_vector_type(8)))  float  v8f;
typedef __attribute__((ext_vector_type(4)))  int    v4i;

static constexpr int BM    = 128;  // block tile rows (M)
static constexpr int BN    = 64;   // block tile cols (N)
static constexpr int KSTEP = 64;   // K per staging slab (2 WMMA substeps of 32)
static constexpr int LDSK  = 80;   // padded K stride in LDS (160B rows)

static constexpr int T = 2048, NN = 2048, D = 512, NH = 8;
static constexpr float SCALE = 0.022097086912079608f;  // 2048^-0.5
static constexpr double TWO_PI = 6.283185307179586476925286766559;

#if defined(__AMDGCN__) && __has_builtin(__builtin_amdgcn_global_load_async_to_lds_b128) && \
    __has_builtin(__builtin_amdgcn_s_wait_asynccnt)
#define HAVE_ASYNC 1
#define ASYNC_WAIT(n) __builtin_amdgcn_s_wait_asynccnt(n)
#else
#define HAVE_ASYNC 0
#define ASYNC_WAIT(n)
#endif

struct __align__(16) Smem {
    __bf16 A[BM * LDSK];
    __bf16 B[BN * LDSK];
};

// 16B global -> LDS copy: async DMA path on gfx1250 (ASYNCcnt), else via VGPRs.
__device__ __forceinline__ void cp16(__bf16* lds_dst, const __bf16* gsrc) {
#if HAVE_ASYNC
    __builtin_amdgcn_global_load_async_to_lds_b128(
        (__attribute__((address_space(1))) v4i*)gsrc,
        (__attribute__((address_space(3))) v4i*)lds_dst, 0, 0);
#else
    *(uint4*)lds_dst = *(const uint4*)gsrc;
#endif
}

// ---------------- staging: global -> LDS (straight-line, fixed trip counts) ----------------
// A row-major (M x K): 128 rows x 64 halfs = 1024 16B chunks -> 4 async b128 per wave.
__device__ __forceinline__ void stage_A_rows(Smem& sm, const __bf16* g, i64 ld, int k0) {
#pragma unroll
    for (int u = 0; u < 4; ++u) {
        int i = (int)threadIdx.x + u * 256;
        int r = i >> 3, c = (i & 7) * 8;
        cp16(&sm.A[r * LDSK + c], &g[(i64)r * ld + k0 + c]);
    }
}
// B operand N-major in LDS; source row-major (N x K): 64 rows -> 2 async b128 per wave.
__device__ __forceinline__ void stage_B_rows(Smem& sm, const __bf16* g, i64 ld, int k0) {
#pragma unroll
    for (int u = 0; u < 2; ++u) {
        int i = (int)threadIdx.x + u * 256;
        int r = i >> 3, c = (i & 7) * 8;
        cp16(&sm.B[r * LDSK + c], &g[(i64)r * ld + k0 + c]);
    }
}
// A = G^T, G row-major (K x M): transpose-scatter (sync path, DScnt).
__device__ __forceinline__ void stage_A_trans(Smem& sm, const __bf16* g, i64 ld, int k0, int mBase) {
#pragma unroll
    for (int u = 0; u < 4; ++u) {
        int i = (int)threadIdx.x + u * 256;           // KSTEP * (BM/8) = 1024
        int kk = i >> 4, mm = (i & 15) * 8;
        v8bf vv = *(const v8bf*)&g[(i64)(k0 + kk) * ld + mBase + mm];
#pragma unroll
        for (int j = 0; j < 8; ++j) sm.A[(mm + j) * LDSK + kk] = vv[j];
    }
}
// B from row-major (K x N): transpose-scatter into N-major LDS.
__device__ __forceinline__ void stage_B_trans(Smem& sm, const __bf16* g, i64 ld, int k0, int nBase) {
#pragma unroll
    for (int u = 0; u < 2; ++u) {
        int i = (int)threadIdx.x + u * 256;           // KSTEP * (BN/8) = 512
        int kk = i >> 3, nn = (i & 7) * 8;
        v8bf vv = *(const v8bf*)&g[(i64)(k0 + kk) * ld + nBase + nn];
#pragma unroll
        for (int j = 0; j < 8; ++j) sm.B[(nn + j) * LDSK + kk] = vv[j];
    }
}

// ---------------- LDS -> WMMA fragments (ISA 16-bit operand layouts) ----------------
__device__ __forceinline__ v16bf ldsfragA(const __bf16* L, int row0, int kc, int lane) {
    const __bf16* p = L + (row0 + (lane & 15)) * LDSK + kc + ((lane >> 4) << 3);
    v8bf lo = *(const v8bf*)p;
    v8bf hi = *(const v8bf*)(p + 16);
    return __builtin_shufflevector(lo, hi, 0,1,2,3,4,5,6,7,8,9,10,11,12,13,14,15);
}
__device__ __forceinline__ v16bf ldsfragB(const __bf16* L, int col0, int kc, int lane) {
    const __bf16* p = L + (col0 + (lane & 15)) * LDSK + kc + ((lane >> 4) << 3);
    v8bf lo = *(const v8bf*)p;
    v8bf hi = *(const v8bf*)(p + 16);
    return __builtin_shufflevector(lo, hi, 0,1,2,3,4,5,6,7,8,9,10,11,12,13,14,15);
}

__device__ __forceinline__ void wmma_substep(const Smem& sm, int kc, int mW, int nW, int lane,
                                             v8f (&acc)[2][2]) {
    v16bf a0 = ldsfragA(sm.A, mW,      kc, lane);
    v16bf a1 = ldsfragA(sm.A, mW + 16, kc, lane);
    v16bf b0 = ldsfragB(sm.B, nW,      kc, lane);
    v16bf b1 = ldsfragB(sm.B, nW + 16, kc, lane);
    acc[0][0] = __builtin_amdgcn_wmma_f32_16x16x32_bf16(false, a0, false, b0, (short)0, acc[0][0], false, false);
    acc[0][1] = __builtin_amdgcn_wmma_f32_16x16x32_bf16(false, a0, false, b1, (short)0, acc[0][1], false, false);
    acc[1][0] = __builtin_amdgcn_wmma_f32_16x16x32_bf16(false, a1, false, b0, (short)0, acc[1][0], false, false);
    acc[1][1] = __builtin_amdgcn_wmma_f32_16x16x32_bf16(false, a1, false, b1, (short)0, acc[1][1], false, false);
}
__device__ __forceinline__ void wmma_step(const Smem& sm, int mW, int nW, int lane, v8f (&acc)[2][2]) {
    wmma_substep(sm, 0,  mW, nW, lane, acc);
    wmma_substep(sm, 32, mW, nW, lane, acc);
}

// ---------------- K1: RoPE + f32 -> bf16 ----------------
__global__ __launch_bounds__(256) void rope_kernel(const float* __restrict__ Q,
                                                   __bf16* __restrict__ QR,
                                                   const int* __restrict__ posOff) {
    int row = blockIdx.x;            // b*nh*T
    int t   = row & (T - 1);
    i64 base = (i64)row * NN;
    double pos = (double)(posOff[0] + t);
    for (int p = threadIdx.x; p < NN / 2; p += 256) {
        double freq = exp2(-(double)p / 64.0) * (1.0 / TWO_PI);   // theta=2^16, N=2048
        double ang  = fmod(pos * freq, 1.0) * TWO_PI;
        float c = (float)cos(ang), s = (float)sin(ang);
        float vr = Q[base + 2 * p], vi = Q[base + 2 * p + 1];
        QR[base + 2 * p]     = (__bf16)(vr * c - vi * s);
        QR[base + 2 * p + 1] = (__bf16)(vr * s + vi * c);
    }
}

// ---------------- K2: f32 -> bf16 convert ----------------
__global__ __launch_bounds__(256) void cvt_kernel(const float* __restrict__ src,
                                                  __bf16* __restrict__ dst, i64 n) {
    i64 i = (i64)blockIdx.x * 256 + threadIdx.x;
    if (i < n) dst[i] = (__bf16)src[i];
}

// One full K sweep of A·B^T with double-buffered async staging; peeled epilogue.
// Inlined twice with distinct accumulator arrays (no runtime acc select).
__device__ __forceinline__ void score_pass(Smem* sm, const __bf16* Ap, const __bf16* Bp,
                                           int mW, int nW, int lane, v8f (&acc)[2][2]) {
    stage_A_rows(sm[0], Ap, NN, 0);   // 6 async b128 per wave per slab
    stage_B_rows(sm[0], Bp, NN, 0);
#pragma unroll 1
    for (int k0 = 0; k0 < NN - KSTEP; k0 += KSTEP) {
        int cur = (k0 >> 6) & 1;
        stage_A_rows(sm[cur ^ 1], Ap, NN, k0 + KSTEP);   // prefetch next slab
        stage_B_rows(sm[cur ^ 1], Bp, NN, k0 + KSTEP);
        ASYNC_WAIT(6);               // in-order: current slab's 6 ops complete
        __syncthreads();
        wmma_step(sm[cur], mW, nW, lane, acc);
        __syncthreads();
    }
    constexpr int curF = ((NN - KSTEP) >> 6) & 1;        // last slab's buffer
    ASYNC_WAIT(0);
    __syncthreads();
    wmma_step(sm[curF], mW, nW, lane, acc);
    __syncthreads();
}

// ---------------- K3: causal diff scores (per b,g) ----------------
__global__ __launch_bounds__(256) void scores_kernel(const __bf16* __restrict__ QR,
                                                     const float* __restrict__ lambda_p,
                                                     __bf16* __restrict__ S, int bg) {
    int b = bg >> 2, g = bg & 3;
    int x = blockIdx.x;
    int ti = 0;
    while ((ti + 1) * (ti + 2) <= x) ++ti;          // tile row; base = ti*(ti+1)
    int si = x - ti * (ti + 1);                     // s tile within causal bound
    int tBase = ti * BM, sBase = si * BN;

    const __bf16* Q1 = QR + (i64)(b * NH + 2 * g) * T * NN;
    const __bf16* Q2 = Q1 + (i64)T * NN;

    __shared__ Smem sm[2];
    int lane = threadIdx.x & 31, wave = threadIdx.x >> 5;
    int mW = (wave >> 1) * 32, nW = (wave & 1) * 32;

    v8f acc1[2][2] = {}, acc2[2][2] = {};
    score_pass(sm, Q1 + (i64)tBase * NN, Q1 + (i64)sBase * NN, mW, nW, lane, acc1);
    score_pass(sm, Q2 + (i64)tBase * NN, Q2 + (i64)sBase * NN, mW, nW, lane, acc2);

    float lam = 1.0f / (1.0f + __expf(-lambda_p[g]));
#pragma unroll
    for (int i = 0; i < 2; ++i)
#pragma unroll
        for (int j = 0; j < 2; ++j)
#pragma unroll
            for (int v = 0; v < 8; ++v) {
                int tG = tBase + mW + i * 16 + v + ((lane >> 4) << 3);
                int sG = sBase + nW + j * 16 + (lane & 15);
                float d = (acc1[i][j][v] - lam * acc2[i][j][v]) * SCALE;
                S[(i64)tG * T + sG] = (__bf16)((sG < tG) ? d : 0.0f);  // strict lower
            }
}

// ---------------- K4: out_g = S @ V (causal K-bound), duplicated to head pair ----------------
__global__ __launch_bounds__(256) void outg_kernel(const __bf16* __restrict__ S,
                                                   const __bf16* __restrict__ Vbf,
                                                   float* __restrict__ out, int bg) {
    int b = bg >> 2, g = bg & 3;
    int tBase = blockIdx.x * BM, dBase = blockIdx.y * BN;
    const __bf16* Ap = S + (i64)tBase * T;
    const __bf16* Bg = Vbf + (i64)b * T * D;

    __shared__ Smem sm;
    int lane = threadIdx.x & 31, wave = threadIdx.x >> 5;
    int mW = (wave >> 1) * 32, nW = (wave & 1) * 32;
    v8f acc[2][2] = {};

    int kEnd = tBase + BM;   // scores are zero / never stored for s >= t
#pragma unroll 1
    for (int k0 = 0; k0 < kEnd; k0 += KSTEP) {
        stage_A_rows(sm, Ap, T, k0);          // async
        stage_B_trans(sm, Bg, D, k0, dBase);  // sync transpose
        ASYNC_WAIT(0);
        __syncthreads();
        wmma_step(sm, mW, nW, lane, acc);
        __syncthreads();
    }
    float* o1 = out + (i64)(b * NH + 2 * g) * T * D;
    float* o2 = o1 + (i64)T * D;
#pragma unroll
    for (int i = 0; i < 2; ++i)
#pragma unroll
        for (int j = 0; j < 2; ++j)
#pragma unroll
            for (int v = 0; v < 8; ++v) {
                int tG = tBase + mW + i * 16 + v + ((lane >> 4) << 3);
                int dG = dBase + nW + j * 16 + (lane & 15);
                float val = acc[i][j][v];
                o1[(i64)tG * D + dG] = val;
                o2[(i64)tG * D + dG] = val;
            }
}

// ---------------- K5: out += QR @ state ----------------
__global__ __launch_bounds__(256) void addqs_kernel(const __bf16* __restrict__ QR,
                                                    const __bf16* __restrict__ Stb,
                                                    float* __restrict__ out) {
    int bh = blockIdx.z;
    int tBase = blockIdx.x * BM, dBase = blockIdx.y * BN;
    const __bf16* Ap = QR + (i64)bh * T * NN + (i64)tBase * NN;
    const __bf16* Bp = Stb + (i64)bh * NN * D;

    __shared__ Smem sm;
    int lane = threadIdx.x & 31, wave = threadIdx.x >> 5;
    int mW = (wave >> 1) * 32, nW = (wave & 1) * 32;
    v8f acc[2][2] = {};
#pragma unroll 1
    for (int k0 = 0; k0 < NN; k0 += KSTEP) {
        stage_A_rows(sm, Ap, NN, k0);
        stage_B_trans(sm, Bp, D, k0, dBase);
        ASYNC_WAIT(0);
        __syncthreads();
        wmma_step(sm, mW, nW, lane, acc);
        __syncthreads();
    }
#pragma unroll
    for (int i = 0; i < 2; ++i)
#pragma unroll
        for (int j = 0; j < 2; ++j)
#pragma unroll
            for (int v = 0; v < 8; ++v) {
                int tG = tBase + mW + i * 16 + v + ((lane >> 4) << 3);
                int dG = dBase + nW + j * 16 + (lane & 15);
                i64 idx = (i64)bh * T * D + (i64)tG * D + dG;
                out[idx] += acc[i][j][v];
            }
}

// ---------------- K6: new_state = state + scale * QR^T @ V ----------------
__global__ __launch_bounds__(256) void nstate_kernel(const __bf16* __restrict__ QR,
                                                     const __bf16* __restrict__ Vbf,
                                                     const float* __restrict__ state,
                                                     float* __restrict__ ns) {
    int bh = blockIdx.z, b = bh >> 3;
    int nBase = blockIdx.x * BM, dBase = blockIdx.y * BN;
    const __bf16* Gq = QR + (i64)bh * T * NN;   // (T x N) row-major; A = Gq^T
    const __bf16* Gv = Vbf + (i64)b * T * D;

    __shared__ Smem sm;
    int lane = threadIdx.x & 31, wave = threadIdx.x >> 5;
    int mW = (wave >> 1) * 32, nW = (wave & 1) * 32;
    v8f acc[2][2] = {};
#pragma unroll 1
    for (int k0 = 0; k0 < T; k0 += KSTEP) {
        stage_A_trans(sm, Gq, NN, k0, nBase);
        stage_B_trans(sm, Gv, D, k0, dBase);
        __syncthreads();
        wmma_step(sm, mW, nW, lane, acc);
        __syncthreads();
    }
#pragma unroll
    for (int i = 0; i < 2; ++i)
#pragma unroll
        for (int j = 0; j < 2; ++j)
#pragma unroll
            for (int v = 0; v < 8; ++v) {
                int nG = nBase + mW + i * 16 + v + ((lane >> 4) << 3);
                int dG = dBase + nW + j * 16 + (lane & 15);
                i64 idx = (i64)bh * NN * D + (i64)nG * D + dG;
                ns[idx] = state[idx] + SCALE * acc[i][j][v];
            }
}

extern "C" void kernel_launch(void* const* d_in, const int* in_sizes, int n_in,
                              void* d_out, int out_size, void* d_ws, size_t ws_size,
                              hipStream_t stream) {
    (void)in_sizes; (void)n_in; (void)out_size; (void)ws_size;
    const float* Q        = (const float*)d_in[0];   // (2,8,2048,2048)
    const float* V        = (const float*)d_in[1];   // (2,1,2048,512)
    const float* state    = (const float*)d_in[2];   // (2,8,2048,512)
    const float* lambda_p = (const float*)d_in[3];   // (1,4,1,1)
    const int*   posOff   = (const int*)d_in[4];

    float* out = (float*)d_out;                          // (2,8,2048,512)
    float* new_state = out + (i64)2 * NH * T * D;        // (2,8,2048,512)

    char* ws = (char*)d_ws;
    __bf16* QRbf = (__bf16*)ws;                                         // 134,217,728 B
    __bf16* Stbf = (__bf16*)(ws + (i64)134217728);                      //  33,554,432 B
    __bf16* Vbf  = (__bf16*)(ws + (i64)134217728 + 33554432);           //   4,194,304 B
    __bf16* S    = (__bf16*)(ws + (i64)134217728 + 33554432 + 4194304); //   8,388,608 B (per-combo)

    rope_kernel<<<2 * NH * T, 256, 0, stream>>>(Q, QRbf, posOff);

    i64 nState = (i64)2 * NH * NN * D;       // 16,777,216
    i64 nV     = (i64)2 * T * D;             //  2,097,152
    cvt_kernel<<<(unsigned)((nState + 255) / 256), 256, 0, stream>>>(state, Stbf, nState);
    cvt_kernel<<<(unsigned)((nV + 255) / 256), 256, 0, stream>>>(V, Vbf, nV);

    for (int bg = 0; bg < 8; ++bg) {
        scores_kernel<<<272, 256, 0, stream>>>(QRbf, lambda_p, S, bg);   // 272 causal tiles
        outg_kernel<<<dim3(T / BM, D / BN), 256, 0, stream>>>(S, Vbf, out, bg);
    }
    addqs_kernel<<<dim3(T / BM, D / BN, 16), 256, 0, stream>>>(QRbf, Stbf, out);
    nstate_kernel<<<dim3(NN / BM, D / BN, 16), 256, 0, stream>>>(QRbf, Vbf, state, new_state);
}
